// STU_16827681865996
// MI455X (gfx1250) — compile-verified, matrix operand order checked by reading, not run on hardware
//
#include <hip/hip_runtime.h>
#include <hip/hip_bf16.h>

// ---------------------------------------------------------------------------
// STU forward for MI455X (gfx1250, wave32, WMMA bf16 16x16x32, f32 accum)
//
//   P_br[b,s,k,o]  = sum_i M_br[k,o,i] * u[b,s,i]                 (GEMM)
//   spec[b,l,o]    = sum_br sum_k sum_{s<=l} f_br[l-s,k] P_br[..] (Toeplitz GEMM)
//       f_plus[t,k]  = phi[t,k]*sigma[k]^0.25
//       f_minus[t,k] = f_plus[t,k]*(-1)^t
//   aru[b,t,o]     = sum_{j<3} sum_i M_u[j,o,i] u[b,t-j,i]        (GEMM)
//   y_t = aru[t] + spec[t-2];  yhat_t = y_t + m0@yhat_{t-1} + m1@yhat_{t-2}
// ---------------------------------------------------------------------------

#define BSZ   2
#define SL    1024
#define KF    24
#define DIN   512
#define DOUT  512
#define KU    3
#define KY    2

typedef __attribute__((ext_vector_type(16))) __bf16 v16bf;
typedef __attribute__((ext_vector_type(8)))  __bf16 v8bf;
typedef __attribute__((ext_vector_type(8)))  float  v8f;

#if defined(__HIP_DEVICE_COMPILE__) && defined(__gfx1250__)
#define USE_ASYNC_LDS 1
#else
#define USE_ASYNC_LDS 0
#endif

#if USE_ASYNC_LDS
// Async global->LDS copy (tracked by ASYNCcnt). VDST = LDS byte offset: for an
// LDS-aperture generic pointer, addr[31:0] is the LDS offset (ISA 10.2).
__device__ __forceinline__ void async_g2l_b128(void* lds_dst, const void* gsrc) {
  unsigned off = (unsigned)(unsigned long long)lds_dst;
  asm volatile("global_load_async_to_lds_b128 %0, %1, off"
               :: "v"(off), "v"(gsrc)
               : "memory");
}
__device__ __forceinline__ void wait_async0() {
#if __has_builtin(__builtin_amdgcn_s_wait_asynccnt)
  __builtin_amdgcn_s_wait_asynccnt(0);
#else
  asm volatile("s_wait_asynccnt 0x0" ::: "memory");
#endif
}
#endif

__device__ __forceinline__ __bf16 f2bf(float f) {
  unsigned u = __builtin_bit_cast(unsigned, f);
  unsigned r = u + 0x7FFFu + ((u >> 16) & 1u);   // round-to-nearest-even
  unsigned short h = (unsigned short)(r >> 16);
  return __builtin_bit_cast(__bf16, h);
}
__device__ __forceinline__ float bf2f(__bf16 b) {
  unsigned short h = __builtin_bit_cast(unsigned short, b);
  unsigned u = ((unsigned)h) << 16;
  return __builtin_bit_cast(float, u);
}
__device__ __forceinline__ v8f wmma_bf16(v16bf a, v16bf b, v8f c) {
  return __builtin_amdgcn_wmma_f32_16x16x32_bf16(false, a, false, b, (short)0, c,
                                                 false, false);
}

// ---------------- prep kernels ------------------------------------------------

__global__ void k_cvt(const float* __restrict__ src, __bf16* __restrict__ dst, int n) {
  int i = blockIdx.x * blockDim.x + threadIdx.x;
  if (i < n) dst[i] = f2bf(src[i]);
}

// G_shift[br][r][k][m], m in [0,2048): G_shift[br][r][k][m] = Gbr[k][m-r]
// Gbr[k][x] = f_br[SL - x, k] for 1<=x<=SL else 0   (time-reversed, zero-padded)
__global__ void k_gshift(const float* __restrict__ phi, const float* __restrict__ sigma,
                         __bf16* __restrict__ G) {
  int idx = blockIdx.x * blockDim.x + threadIdx.x;
  const int total = 2 * 16 * KF * 2048;
  if (idx >= total) return;
  int m  = idx & 2047;
  int k  = (idx >> 11) % KF;
  int r  = ((idx >> 11) / KF) & 15;
  int br = idx / (2048 * KF * 16);
  float val = 0.f;
  int mm = m - r;
  if (mm >= 1 && mm <= SL) {
    int t = SL - mm;                          // lag 0..SL-1
    float sc = sqrtf(sqrtf(fmaxf(sigma[k], 0.f)));
    val = phi[t * KF + k] * sc;
    if (br == 1 && (t & 1)) val = -val;       // (-1)^t modulation
  }
  G[idx] = f2bf(val);
}

// myT[lag][j][o] = m_y[lag][o][j]  (bf16, o contiguous for coalesced reads)
__global__ void k_myT(const float* __restrict__ my, __bf16* __restrict__ myT, int n) {
  int idx = blockIdx.x * blockDim.x + threadIdx.x;
  if (idx >= n) return;
  int o = idx & (DOUT - 1);
  int j = (idx >> 9) & (DOUT - 1);
  int lag = idx >> 18;
  myT[idx] = f2bf(my[((size_t)lag * DOUT + o) * DOUT + j]);
}

// ---------------- stage P: Pt[br][b][k][o][s] = sum_i M_br[k,o,i] u[b,s,i] ----

__global__ void k_proj(const __bf16* __restrict__ ub, const __bf16* __restrict__ Mb,
                       __bf16* __restrict__ Pt) {
  int wave = (blockIdx.x * (blockDim.x >> 5)) + (threadIdx.x >> 5);
  int lane = threadIdx.x & 31;
  // tiles: [br][b][k][st(64)][ot(32)] = 196608 waves
  if (wave >= 2 * BSZ * KF * 64 * 32) return;
  int ot = wave & 31;
  int st = (wave >> 5) & 63;
  int r  = wave >> 11;           // 0..95
  int k  = r % KF;
  int rb = r / KF;               // 0..3
  int b  = rb & 1;
  int br = rb >> 1;
  int S0 = st * 16, O0 = ot * 16;
  int lrow = lane & 15, hi = lane >> 4;

  const __bf16* arow = ub + ((size_t)b * SL + (S0 + lrow)) * DIN;               // A row = s'
  const __bf16* brow = Mb + (((size_t)br * KF + k) * DOUT + (O0 + lrow)) * DIN; // B col = o'
  v8f acc = {};
#pragma unroll 4
  for (int i0 = 0; i0 < DIN; i0 += 32) {
    v16bf a  = *(const v16bf*)(arow + i0 + hi * 16);
    v16bf bb = *(const v16bf*)(brow + i0 + hi * 16);
    acc = wmma_bf16(a, bb, acc);
  }
  // D element (M=s', N=o'): s = S0 + hi*8 + rr, o = O0 + lrow -> store transposed
  __bf16* out = Pt + ((((size_t)br * BSZ + b) * KF + k) * DOUT + (O0 + lrow)) * SL
                   + S0 + hi * 8;
  v8bf ov;
#pragma unroll
  for (int rr = 0; rr < 8; ++rr) ov[rr] = f2bf(acc[rr]);
  *(v8bf*)out = ov;
}

// ---------------- ar_u: aru[b,t,o] = sum_j sum_i Mu[j,o,i] u[b,t-j,i] --------

__global__ void k_aru(const __bf16* __restrict__ ub, const __bf16* __restrict__ Mub,
                      float* __restrict__ aru) {
  int wave = (blockIdx.x * (blockDim.x >> 5)) + (threadIdx.x >> 5);
  int lane = threadIdx.x & 31;
  if (wave >= BSZ * 64 * 32) return;         // [b][tt(64)][ot(32)]
  int ot = wave & 31;
  int tt = (wave >> 5) & 63;
  int b  = wave >> 11;
  int T0 = tt * 16, O0 = ot * 16;
  int lrow = lane & 15, hi = lane >> 4;

  v8f acc = {};
  for (int j = 0; j < KU; ++j) {
    const __bf16* arow = Mub + ((size_t)j * DOUT + (O0 + lrow)) * DIN;  // A row = o'
    int trow = T0 + lrow - j;                                           // B col = t'
    const __bf16* bbase = ub + ((size_t)b * SL + trow) * DIN;
    bool valid = (trow >= 0);
#pragma unroll 4
    for (int i0 = 0; i0 < DIN; i0 += 32) {
      v16bf a = *(const v16bf*)(arow + i0 + hi * 16);
      v16bf bb;
#pragma unroll
      for (int q = 0; q < 16; ++q) bb[q] = f2bf(0.f);
      if (valid) bb = *(const v16bf*)(bbase + i0 + hi * 16);
      acc = wmma_bf16(a, bb, acc);
    }
  }
  // D element (M=o', N=t'): o = O0 + hi*8 + rr, t = T0 + lrow (contig in o)
  float* out = aru + ((size_t)b * SL + (T0 + lrow)) * DOUT + O0 + hi * 8;
  *(v8f*)out = acc;
}

// ---------------- Toeplitz conv: spec[b,l,o] -----------------------------------
// All 8 waves of a block share (b, l-tile) => identical G window; stage it into
// LDS with CDNA5 async global->LDS copies (ASYNCcnt), B operand read from LDS.

#define GSTRIDE 1040   // bf16 elements per staged row (2080 B, 32B aligned)

__global__ void k_conv(const __bf16* __restrict__ Pt, const __bf16* __restrict__ G,
                       float* __restrict__ spec) {
  __shared__ __align__(32) __bf16 gbuf[16 * GSTRIDE];   // 33 KB
  int wave = (blockIdx.x * (blockDim.x >> 5)) + (threadIdx.x >> 5);
  int lane = threadIdx.x & 31;
  // [b][lt(64)][og(8)] = 1024 waves, 8 per block (all share b, lt)
  int og = wave & 7;
  int lt = (wave >> 3) & 63;
  int b  = wave >> 9;
  int L0 = lt * 16, O0 = og * 64;
  int lrow = lane & 15, hi = lane >> 4;
  int nS = lt / 2 + 1;            // s-tiles of 32, covering s <= L0+15
  int W  = nS * 32;               // staged window (elements), multiple of 32
  int startm = SL - L0;           // multiple of 16

  v8f acc0 = {}, acc1 = {}, acc2 = {}, acc3 = {};
  for (int br = 0; br < 2; ++br) {
    for (int k = 0; k < KF; ++k) {
      __syncthreads();            // previous window fully consumed
      {
        int chunksPerRow = W >> 3;              // 16B chunks per row
        int total = chunksPerRow << 4;          // 16 rows
        for (int c = threadIdx.x; c < total; c += blockDim.x) {
          int rr = c / chunksPerRow;
          int cc = c - rr * chunksPerRow;
          const __bf16* src = G + (((size_t)br * 16 + rr) * KF + k) * 2048
                                + startm + cc * 8;
          __bf16* dst = gbuf + rr * GSTRIDE + cc * 8;
#if USE_ASYNC_LDS
          async_g2l_b128(dst, src);
#else
          *(v8bf*)dst = *(const v8bf*)src;
#endif
        }
#if USE_ASYNC_LDS
        wait_async0();                          // drain this wave's async copies
#endif
      }
      __syncthreads();            // LDS window visible to all waves

      // B operand (Toeplitz): lane col l' = lrow, reduction s' = hi*16 + j
      const __bf16* gb = gbuf + lrow * GSTRIDE + hi * 16;
      const __bf16* Abase = Pt + ((((size_t)br * BSZ + b) * KF + k) * DOUT
                             + (O0 + lrow)) * SL + hi * 16;
      for (int ss = 0; ss < nS; ++ss) {
        int S0 = ss * 32;
        v16bf bb = *(const v16bf*)(gb + S0);
        const __bf16* ap = Abase + S0;
        v16bf a0 = *(const v16bf*)(ap);
        v16bf a1 = *(const v16bf*)(ap + 16 * SL);
        v16bf a2 = *(const v16bf*)(ap + 32 * SL);
        v16bf a3 = *(const v16bf*)(ap + 48 * SL);
        __builtin_prefetch(ap + 32, 0, 1);      // global_prefetch next s-chunk
        acc0 = wmma_bf16(a0, bb, acc0);
        acc1 = wmma_bf16(a1, bb, acc1);
        acc2 = wmma_bf16(a2, bb, acc2);
        acc3 = wmma_bf16(a3, bb, acc3);
      }
    }
  }
  // D element (M=o', N=l'): o = O0 + 16q + hi*8 + rr, l = L0 + lrow (contig in o)
  float* out = spec + ((size_t)b * SL + (L0 + lrow)) * DOUT + O0 + hi * 8;
  *(v8f*)(out)      = acc0;
  *(v8f*)(out + 16) = acc1;
  *(v8f*)(out + 32) = acc2;
  *(v8f*)(out + 48) = acc3;
}

// ---------------- AR(2) output recurrence (sequential scan) -------------------

__global__ void __launch_bounds__(1024) k_rec(const float* __restrict__ aru,
                                              const float* __restrict__ spec,
                                              const __bf16* __restrict__ myT,
                                              float* __restrict__ out) {
  __shared__ float c1[BSZ * DOUT];
  __shared__ float c2[BSZ * DOUT];
  int tid = threadIdx.x;
  int b = tid >> 9;
  int o = tid & (DOUT - 1);
  c1[tid] = 0.f;
  c2[tid] = 0.f;
  __syncthreads();
  const __bf16* m0 = myT + o;                 // myT[0][j][o], stride DOUT
  const __bf16* m1 = myT + (size_t)DOUT * DOUT + o;
  for (int t = 0; t < SL; ++t) {
    float y = aru[((size_t)b * SL + t) * DOUT + o];
    if (t >= KY) y += spec[((size_t)b * SL + (t - KY)) * DOUT + o];
    const float* cc1 = c1 + b * DOUT;
    const float* cc2 = c2 + b * DOUT;
    float ar = 0.f;
#pragma unroll 8
    for (int j = 0; j < DOUT; ++j) {
      ar += bf2f(m0[(size_t)j * DOUT]) * cc1[j] + bf2f(m1[(size_t)j * DOUT]) * cc2[j];
    }
    float yh = y + ar;
    out[((size_t)b * SL + t) * DOUT + o] = yh;
    __syncthreads();
    float oc1 = c1[tid];
    c2[tid] = oc1;
    c1[tid] = yh;
    __syncthreads();
  }
}

// ---------------- host side ----------------------------------------------------

extern "C" void kernel_launch(void* const* d_in, const int* in_sizes, int n_in,
                              void* d_out, int out_size, void* d_ws, size_t ws_size,
                              hipStream_t stream) {
  (void)in_sizes; (void)n_in; (void)out_size; (void)ws_size;
  const float* u      = (const float*)d_in[0];   // (B, SL, DIN)
  const float* M_u    = (const float*)d_in[1];   // (KU, DOUT, DIN)
  const float* M_plus = (const float*)d_in[2];   // (KF, DOUT, DIN)
  const float* M_minus= (const float*)d_in[3];   // (KF, DOUT, DIN)
  const float* m_y    = (const float*)d_in[4];   // (KY, DOUT, DOUT)
  const float* sigma  = (const float*)d_in[5];   // (KF,)
  const float* phi    = (const float*)d_in[6];   // (SL, KF)
  float* yout = (float*)d_out;

  char* ws = (char*)d_ws;
  const size_t off_ub   = 0;                                    // bf16 u
  const size_t off_G    = off_ub   + (size_t)BSZ*SL*DIN*2;      // G_shift
  const size_t off_Mb   = off_G    + (size_t)2*16*KF*2048*2;    // bf16 M_plus|M_minus
  const size_t off_Mub  = off_Mb   + (size_t)2*KF*DOUT*DIN*2;   // bf16 M_u
  const size_t off_Pt   = off_Mub  + (size_t)KU*DOUT*DIN*2;     // Pt (96 MB)
  const size_t off_spec = off_Pt   + (size_t)2*BSZ*KF*DOUT*SL*2;
  const size_t off_aru  = off_spec + (size_t)BSZ*SL*DOUT*4;
  const size_t off_myT  = off_aru  + (size_t)BSZ*SL*DOUT*4;
  __bf16* ub   = (__bf16*)(ws + off_ub);
  __bf16* G    = (__bf16*)(ws + off_G);
  __bf16* Mb   = (__bf16*)(ws + off_Mb);
  __bf16* Mub  = (__bf16*)(ws + off_Mub);
  __bf16* Pt   = (__bf16*)(ws + off_Pt);
  float*  spec = (float*)(ws + off_spec);
  float*  aru  = (float*)(ws + off_aru);
  __bf16* myT  = (__bf16*)(ws + off_myT);

  {
    int n = BSZ * SL * DIN;
    k_cvt<<<(n + 255) / 256, 256, 0, stream>>>(u, ub, n);
  }
  {
    int n = KF * DOUT * DIN;                            // M_plus -> Mb[0]
    k_cvt<<<(n + 255) / 256, 256, 0, stream>>>(M_plus, Mb, n);
    k_cvt<<<(n + 255) / 256, 256, 0, stream>>>(M_minus, Mb + (size_t)n, n);
  }
  {
    int n = KU * DOUT * DIN;
    k_cvt<<<(n + 255) / 256, 256, 0, stream>>>(M_u, Mub, n);
  }
  {
    int n = 2 * 16 * KF * 2048;
    k_gshift<<<(n + 255) / 256, 256, 0, stream>>>(phi, sigma, G);
  }
  {
    int n = KY * DOUT * DOUT;
    k_myT<<<(n + 255) / 256, 256, 0, stream>>>(m_y, myT, n);
  }
  {
    int waves = 2 * BSZ * KF * 64 * 32;                  // 196608
    k_proj<<<waves / 8, 256, 0, stream>>>(ub, Mb, Pt);
  }
  {
    int waves = BSZ * 64 * 32;                           // 4096
    k_aru<<<waves / 8, 256, 0, stream>>>(ub, Mub, aru);
  }
  {
    int waves = BSZ * 64 * 8;                            // 1024
    k_conv<<<waves / 8, 256, 0, stream>>>(Pt, G, spec);
  }
  k_rec<<<1, 1024, 0, stream>>>(aru, spec, myT, yout);
}
